// EnvironmentEncoder_76158360093051
// MI455X (gfx1250) — matrix operand order repaired and checked by reading, not compiled, not run on hardware
//
#include <hip/hip_runtime.h>
#include <hip/hip_fp16.h>

typedef __attribute__((ext_vector_type(16))) _Float16 v16h;
typedef __attribute__((ext_vector_type(8)))  float    v8f;

#define EMBED_DIM 16
#define NUM_RBF   8
#define TILES_PER_WAVE 4                  // 64 edges per wave

// h[n][d] = embed[elem_idx[n]][d]
__global__ void init_h_kernel(const int* __restrict__ elem_idx,
                              const float* __restrict__ embed,
                              float* __restrict__ h, int N) {
  int i = blockIdx.x * blockDim.x + threadIdx.x;
  if (i < N * EMBED_DIM) {
    int n = i >> 4, d = i & 15;
    h[i] = embed[elem_idx[n] * EMBED_DIM + d];
  }
}

__global__ void copy_h_kernel(const float4* __restrict__ src,
                              float4* __restrict__ dst, int n4) {
  int i = blockIdx.x * blockDim.x + threadIdx.x;
  if (i < n4) dst[i] = src[i];
}

// Each wave processes TILES_PER_WAVE tiles of 16 edges. Per tile: two
// v_wmma_f32_16x16x32_f16 compute W = silu(rbf@W1+b1)@W2+b2; D-matrix layout
// (lane = dim, vgpr = edge) gives coalesced gather of h[tgt] and coalesced
// f32 scatter-atomics into h_out[src]. Weights/biases are hoisted and live
// across the tile loop.
__global__ __launch_bounds__(256) void edge_layer_kernel(
    const int* __restrict__ edge_index,   // [2, E]: row0=src, row1=tgt
    const float* __restrict__ distances,  // [E]
    const float* __restrict__ W1,         // [L, 8, 16]
    const float* __restrict__ b1,         // [L, 16]
    const float* __restrict__ W2,         // [L, 16, 16]
    const float* __restrict__ b2,         // [L, 16]
    const float* __restrict__ h_in,       // [N, 16]
    float* __restrict__ h_out,            // [N, 16] (pre-loaded with h_in copy)
    int E, int layer)
{
  __shared__ float tile[8][16][17];       // per-wave 16x16 silu staging (padded)
  const int lane = threadIdx.x & 31;
  const int wid  = threadIdx.x >> 5;
  const int n    = lane & 15;             // column / embed dim
  const int half = lane >> 4;             // 0: lanes 0-15, 1: lanes 16-31
  const int base0 = (blockIdx.x * 8 + wid) * (16 * TILES_PER_WAVE);
  if (base0 >= E) return;

  const float r_min = 0.94486299430f;     // 0.5 * 1.8897259886
  const float r_max = 7.56f;
  const float cstep = (r_max - r_min) / 7.0f;   // linspace spacing (8 pts)
  const float inv_w = 8.0f / (r_max - r_min);   // 1 / width

  const int* src_idx = edge_index;
  const int* tgt_idx = edge_index + E;

  // ---- loop-invariant: B1, B2 (f16, padded to K=32), biases as C ----
  const float* W1l = W1 + layer * NUM_RBF * EMBED_DIM;
  const float* W2l = W2 + layer * EMBED_DIM * EMBED_DIM;
  v16h bm1 = {};                          // lanes 16-31 = K 8..15, 24..31 -> zero
  v16h bm2 = {};
  if (half == 0) {
    #pragma unroll
    for (int k = 0; k < NUM_RBF; ++k)   bm1[k] = (_Float16)W1l[k * EMBED_DIM + n];
  }
  #pragma unroll
  for (int k = 0; k < 8; ++k) bm2[k] = (_Float16)W2l[(8 * half + k) * EMBED_DIM + n];
  const float b1v = b1[layer * EMBED_DIM + n];
  const float b2v = b2[layer * EMBED_DIM + n];
  v8f c1, c2;
  #pragma unroll
  for (int v = 0; v < 8; ++v) { c1[v] = b1v; c2[v] = b2v; }

  for (int t = 0; t < TILES_PER_WAVE; ++t) {
    const int base = base0 + 16 * t;
    if (base >= E) break;
    const bool full = (base + 16 <= E);

    // ---- edge indices: lane n holds src/tgt of edge base+n (coalesced) ----
    int eL = base + n;
    bool inL = full || (eL < E);
    int sL = inL ? src_idx[eL] : 0;
    int tL = inL ? tgt_idx[eL] : 0;

    // ---- A1: rbf tile, rows = 16 edges, K = 8 (zero-padded to 32) ----
    float d = inL ? distances[eL] : 0.0f;
    v16h a1 = {};                         // lanes 16-31 hold K=8..15 -> zero
    if (half == 0) {
      #pragma unroll
      for (int k = 0; k < NUM_RBF; ++k) {
        float u = (d - (r_min + cstep * (float)k)) * inv_w;
        float r = __expf(-0.5f * u * u);
        a1[k] = (_Float16)(inL ? r : 0.0f);
      }
    }

    v8f d1 = __builtin_amdgcn_wmma_f32_16x16x32_f16(false, a1, false, bm1,
                                                    (short)0, c1, false, false);

    // ---- SiLU (x * v_rcp_f32(1+exp(-x))), stage via LDS: D- -> A-layout ----
    #pragma unroll
    for (int v = 0; v < 8; ++v) {
      float x = d1[v];                    // element (M = v + 8*half, N = n)
      tile[wid][v + 8 * half][n] = x * __builtin_amdgcn_rcpf(1.0f + __expf(-x));
    }

    // ---- A2: hidden 16x16 (K=16, padded to 32) ----
    v16h a2 = {};                         // elems 8..15 = K 16..31 -> zero
    #pragma unroll
    for (int k = 0; k < 8; ++k)
      a2[k] = (_Float16)tile[wid][n][8 * half + k];

    v8f d2 = __builtin_amdgcn_wmma_f32_16x16x32_f16(false, a2, false, bm2,
                                                    (short)0, c2, false, false);

    // ---- messages = h[tgt] * W ; scatter-add into h_out[src] ----
    // D2 element v on this lane belongs to edge base + v + 8*half; its
    // src/tgt come from lane (v + 8*half) via ds_bpermute (no VMEM traffic).
    if (full) {                           // fast path: no EXEC masking
      #pragma unroll
      for (int v = 0; v < 8; ++v) {
        int tt = __shfl(tL, v + 8 * half, 32);
        int ss = __shfl(sL, v + 8 * half, 32);
        float msg = h_in[tt * EMBED_DIM + n] * d2[v]; // 16 lanes = 1 line
        unsafeAtomicAdd(&h_out[ss * EMBED_DIM + n], msg);
      }
    } else {                              // guarded tail tile
      #pragma unroll
      for (int v = 0; v < 8; ++v) {
        int e = base + v + 8 * half;
        int tt = __shfl(tL, v + 8 * half, 32);
        int ss = __shfl(sL, v + 8 * half, 32);
        if (e < E) {
          float msg = h_in[tt * EMBED_DIM + n] * d2[v];
          unsafeAtomicAdd(&h_out[ss * EMBED_DIM + n], msg);
        }
      }
    }
  }
}

extern "C" void kernel_launch(void* const* d_in, const int* in_sizes, int n_in,
                              void* d_out, int out_size, void* d_ws, size_t ws_size,
                              hipStream_t stream) {
  const int*   elem_idx   = (const int*)d_in[0];
  const int*   edge_index = (const int*)d_in[1];
  const float* distances  = (const float*)d_in[2];
  const float* embed      = (const float*)d_in[3];
  const float* W1         = (const float*)d_in[4];
  const float* b1         = (const float*)d_in[5];
  const float* W2         = (const float*)d_in[6];
  const float* b2         = (const float*)d_in[7];

  float* hOut = (float*)d_out;
  float* hA   = (float*)d_ws;             // needs N*16*4 = 6.4 MB scratch

  const int N = in_sizes[0];
  const int E = in_sizes[2];

  const int tot = N * EMBED_DIM;
  const int n4  = tot / 4;
  init_h_kernel<<<(tot + 255) / 256, 256, 0, stream>>>(elem_idx, embed, hA, N);

  const int edges_per_block = 8 * 16 * TILES_PER_WAVE;  // 512
  const int egrid = (E + edges_per_block - 1) / edges_per_block;
  const int cgrid = (n4 + 255) / 256;

  // layer 0: out = hA + aggregate(gather from hA)
  copy_h_kernel<<<cgrid, 256, 0, stream>>>((const float4*)hA, (float4*)hOut, n4);
  edge_layer_kernel<<<egrid, 256, 0, stream>>>(edge_index, distances,
                                               W1, b1, W2, b2, hA, hOut, E, 0);
  // layer 1: stable copy of h1 in hA; accumulate into hOut (already = h1)
  copy_h_kernel<<<cgrid, 256, 0, stream>>>((const float4*)hOut, (float4*)hA, n4);
  edge_layer_kernel<<<egrid, 256, 0, stream>>>(edge_index, distances,
                                               W1, b1, W2, b2, hA, hOut, E, 1);
}
